// BilinearAttention_30777735643476
// MI455X (gfx1250) — matrix operand order repaired and checked by reading, not compiled, not run on hardware
//
#include <hip/hip_runtime.h>

typedef _Float16 f16;
typedef __attribute__((ext_vector_type(16))) _Float16 v16h;
typedef __attribute__((ext_vector_type(8)))  _Float16 h8;
typedef __attribute__((ext_vector_type(8)))  float    v8f;
typedef __attribute__((ext_vector_type(4)))  float    f4;
typedef unsigned v4u __attribute__((ext_vector_type(4)));
typedef int      v4i __attribute__((ext_vector_type(4)));
typedef int      v8i __attribute__((ext_vector_type(8)));

static constexpr int Bn = 4, Ln = 2048, Hn = 8, En = 64;
// padded LDS row strides in f16 units (row + 16B pad -> bank-conflict-free)
static constexpr int KROW = 72;   // 32 keys x (64 f16 + 8 pad) = 144B rows
static constexpr int VROW = 40;   // 64 d    x (32 f16 + 8 pad) =  80B rows

static __device__ __forceinline__ v8f wmma16(v16h a, v16h b, v8f c) {
  return __builtin_amdgcn_wmma_f32_16x16x32_f16(false, a, false, b, (short)0, c, false, false);
}

// combine two contiguous 8xf16 (16B) loads into one 16xf16 WMMA operand
static __device__ __forceinline__ v16h ld2(const f16* p0, const f16* p1) {
  h8 lo = *(const h8*)p0;
  h8 hi = *(const h8*)p1;
  return __builtin_shufflevector(lo, hi, 0,1,2,3,4,5,6,7,8,9,10,11,12,13,14,15);
}

static __device__ __forceinline__ unsigned pk16(float a, float b) {
  union { f16 h[2]; unsigned u; } t;
  t.h[0] = (f16)a; t.h[1] = (f16)b;
  return t.u;
}

// ---- Tensor Data Mover: 2D tile (tile_d1 rows of tile_d0 f16) -> LDS, padded rows ----
// D# bitfields per CDNA5 ISA. data_size=1 (2B elements).
static __device__ __forceinline__ void tdm_load_2d(unsigned lds_off, const f16* gptr,
                                                   unsigned tile_d0, unsigned tile_d1,
                                                   unsigned td0, unsigned td1,
                                                   unsigned long long stride_elems,
                                                   unsigned pad_interval, unsigned pad_amount) {
  unsigned long long ga = (unsigned long long)(uintptr_t)gptr;
  v4u g0;
  g0.x = 1u;                                                  // count=1 (valid), user mode
  g0.y = lds_off;                                             // lds_addr [63:32]
  g0.z = (unsigned)ga;                                        // global_addr [95:64]
  g0.w = (unsigned)((ga >> 32) & 0x1FFFFFFu) | (2u << 30);    // global_addr hi + type=2
  unsigned w0 = (1u << 16)                                    // data_size = 2 bytes
              | (1u << 20)                                    // pad_enable
              | (pad_interval << 22) | (pad_amount << 25);
  v8i g1;
  g1[0] = (int)w0;
  g1[1] = (int)((td0 & 0xFFFFu) << 16);                                 // tensor_dim0 lo16
  g1[2] = (int)(((td0 >> 16) & 0xFFFFu) | ((td1 & 0xFFFFu) << 16));     // dim0 hi | dim1 lo
  g1[3] = (int)(((td1 >> 16) & 0xFFFFu) | ((tile_d0 & 0xFFFFu) << 16)); // dim1 hi | tile_dim0
  g1[4] = (int)(tile_d1 & 0xFFFFu);                                     // tile_dim1 (tile_dim2=0)
  g1[5] = (int)(unsigned)(stride_elems & 0xFFFFFFFFu);                  // dim0_stride lo32
  g1[6] = (int)(unsigned)((stride_elems >> 32) & 0xFFFFu);              // stride hi16
  g1[7] = 0;
  v4i z4 = {0, 0, 0, 0};
#if __clang_major__ >= 23
  v8i z8 = {0, 0, 0, 0, 0, 0, 0, 0};
  __builtin_amdgcn_tensor_load_to_lds(g0, g1, z4, z4, z8, 0);
#else
  __builtin_amdgcn_tensor_load_to_lds(g0, g1, z4, z4, 0);
#endif
}

// ---------------- kernel 1: Qh[b,h,l,f] = scale * (Q @ W), f16 ----------------
__global__ __launch_bounds__(256) void qw_kernel(const float* __restrict__ Q,
                                                 const float* __restrict__ W,
                                                 f16* __restrict__ Qh) {
  __shared__ float Wl[64 * 64];
  for (int i = threadIdx.x; i < 4096; i += 256) Wl[i] = W[i];
  __syncthreads();
  int f   = threadIdx.x & 63;
  int row = blockIdx.x * 4 + (threadIdx.x >> 6);   // row = (b*L + l)*H + h
  const float* q = Q + row * 64;
  float s = 0.f;
#pragma unroll
  for (int e = 0; e < 64; ++e) s += q[e] * Wl[e * 64 + f];
  int h = row & 7, l = (row >> 3) & 2047, b = row >> 14;
  Qh[(((b * Hn + h) * Ln + l) << 6) + f] = (f16)(s * 0.125f);   // 1/sqrt(64)
}

// ------- kernel 2: Kh[b,h,s,f] = f16(K);  Vt[b,h,d,s] = f16(V) (transposed) -------
__global__ __launch_bounds__(256) void kv_kernel(const float* __restrict__ K,
                                                 const float* __restrict__ V,
                                                 f16* __restrict__ Kh,
                                                 f16* __restrict__ Vt) {
  int i = blockIdx.x * 256 + threadIdx.x;          // [b][s][h][f]
  int f = i & 63, h = (i >> 6) & 7, s = (i >> 9) & 2047, b = i >> 20;
  Kh[(((b * Hn + h) * Ln + s) << 6) + f] = (f16)K[i];
  Vt[((b * Hn + h) * 64 + f) * Ln + s]   = (f16)V[i];
}

// ------- kernel 3: fused causal flash attention -------
// 4 waves/block, 32 queries/wave (two 16-query groups sharing all A-operands).
// K/V chunks staged in LDS by the Tensor Data Mover, double-buffered.
__global__ __launch_bounds__(128) void fa_kernel(const f16* __restrict__ Qh,
                                                 const f16* __restrict__ Kh,
                                                 const f16* __restrict__ Vt,
                                                 float* __restrict__ out) {
  __shared__ alignas(16) f16 Kl[2][32 * KROW];
  __shared__ alignas(16) f16 Vl[2][64 * VROW];

  const int blk  = blockIdx.x;                 // 0 .. (L/128)*B*H - 1
  const int qblk = 15 - (blk >> 5);            // heavy causal tiles launch first
  const int bh   = blk & 31;
  const int h    = bh & 7;
  const int b    = bh >> 3;
  const int wid  = __builtin_amdgcn_readfirstlane(threadIdx.x >> 5);   // scalar wave id
  const int lane = threadIdx.x & 31;
  const int lo16 = lane & 15;
  const bool low = lane < 16;

  const int q0b = qblk * 128;                  // block's first query
  const int qg0 = q0b + wid * 32;              // this wave's first query (mult of 32)
  const int qa  = qg0 + lo16;                  // group a query
  const int qb  = qg0 + 16 + lo16;             // group b query

  const f16* Qb = Qh + (((size_t)(b * Hn + h) * Ln) << 6);
  const f16* Kg = Kh + (((size_t)(b * Hn + h) * Ln) << 6);
  const f16* Vg = Vt + ((size_t)(b * Hn + h) * 64) * Ln;

  // B operands (qw^T): col = query = lane%16; K(=f) 0-7,16-23 low lanes / 8-15,24-31 high lanes
  const int  qf    = low ? 0 : 8;
  const f16* qrowa = Qb + ((size_t)qa << 6);
  const f16* qrowb = Qb + ((size_t)qb << 6);
  v16h bq0a = ld2(qrowa + qf,      qrowa + qf + 16);
  v16h bq1a = ld2(qrowa + qf + 32, qrowa + qf + 48);
  v16h bq0b = ld2(qrowb + qf,      qrowb + qf + 16);
  v16h bq1b = ld2(qrowb + qf + 32, qrowb + qf + 48);

  v8f accA[4], accB[4];
#pragma unroll
  for (int t = 0; t < 4; ++t) { accA[t] = (v8f){0,0,0,0,0,0,0,0}; accB[t] = (v8f){0,0,0,0,0,0,0,0}; }
  float mA = -1e30f, lA = 0.f, mB = -1e30f, lB = 0.f;

  const int nfull = qg0 >> 5;                  // shared diagonal chunk for both groups
  const int cmax  = (q0b + 96) >> 5;           // last chunk needed by wave 3
  const int fb = low ? 0 : 8;
  const int sb = low ? 0 : 8;

  // prologue: TDM chunk 0 into buffer 0 (wave0: K, wave1: V)
  if (wid == 0) tdm_load_2d((unsigned)(size_t)&Kl[0][0], Kg, 64, 32, 64, 2048, 64, 4, 3);
  if (wid == 1) tdm_load_2d((unsigned)(size_t)&Vl[0][0], Vg, 32, 64, 2048, 64, (unsigned long long)Ln, 3, 3);
  if (wid < 2) __builtin_amdgcn_s_wait_tensorcnt(0);
  __syncthreads();

  for (int c = 0; c <= cmax; ++c) {
    const int cb = c & 1;
    // issue next chunk's DMA into the other buffer (overlaps with compute below)
    if (c < cmax) {
      const int s0n = (c + 1) << 5;
      if (wid == 0) tdm_load_2d((unsigned)(size_t)&Kl[cb ^ 1][0], Kg + ((size_t)s0n << 6),
                                64, 32, 64, 2048, 64, 4, 3);
      if (wid == 1) tdm_load_2d((unsigned)(size_t)&Vl[cb ^ 1][0], Vg + s0n,
                                32, 64, 2048, 64, (unsigned long long)Ln, 3, 3);
    }

    if (c <= nfull) {                          // uniform scalar guard (wave-level)
      const int s0 = c << 5;
      const f16* Kc = &Kl[cb][0];
      const f16* Vc = &Vl[cb][0];

      // ---- shared A operands: K tiles (used by both query groups) ----
      const f16* kr0 = Kc + lo16 * KROW;
      const f16* kr1 = Kc + (16 + lo16) * KROW;
      v16h ak00 = ld2(kr0 + fb,      kr0 + fb + 16);
      v16h ak01 = ld2(kr0 + fb + 32, kr0 + fb + 48);
      v16h ak10 = ld2(kr1 + fb,      kr1 + fb + 16);
      v16h ak11 = ld2(kr1 + fb + 32, kr1 + fb + 48);

      // ---- scores^T for both groups: C[key][query] ----
      v8f c0a = (v8f){0,0,0,0,0,0,0,0}, c1a = (v8f){0,0,0,0,0,0,0,0};
      v8f c0b = (v8f){0,0,0,0,0,0,0,0}, c1b = (v8f){0,0,0,0,0,0,0,0};
      c0a = wmma16(ak00, bq0a, c0a); c0a = wmma16(ak01, bq1a, c0a);
      c1a = wmma16(ak10, bq0a, c1a); c1a = wmma16(ak11, bq1a, c1a);
      c0b = wmma16(ak00, bq0b, c0b); c0b = wmma16(ak01, bq1b, c0b);
      c1b = wmma16(ak10, bq0b, c1b); c1b = wmma16(ak11, bq1b, c1b);

      if (c == nfull) {                        // causal mask: key > query -> -inf
        const int sB = s0 + sb;
#pragma unroll
        for (int r = 0; r < 8; ++r) {
          if (sB + r      > qa) c0a[r] = -1e30f;
          if (sB + 16 + r > qa) c1a[r] = -1e30f;
          if (sB + 16 + r > qb) c1b[r] = -1e30f;   // group b: first tile never masked
        }
      }

      // ---- online softmax + P^T build, per group ----
      union { unsigned u[8]; v16h v; } bpa, bpb;
#pragma unroll
      for (int g = 0; g < 2; ++g) {
        v8f&   c0 = g ? c0b : c0a;
        v8f&   c1 = g ? c1b : c1a;
        float& mr = g ? mB : mA;
        float& lr = g ? lB : lA;
        v8f*   ac = g ? accB : accA;
        auto&  bp = g ? bpb : bpa;

        float mx = c0[0];
#pragma unroll
        for (int r = 1; r < 8; ++r) mx = fmaxf(mx, c0[r]);
#pragma unroll
        for (int r = 0; r < 8; ++r) mx = fmaxf(mx, c1[r]);
        mx = fmaxf(mx, __shfl_xor(mx, 16, 32));
        const float mnew  = fmaxf(mr, mx);
        const float alpha = __expf(mr - mnew);

        float p0[8], p1[8], lsum = 0.f;
#pragma unroll
        for (int r = 0; r < 8; ++r) { p0[r] = __expf(c0[r] - mnew); lsum += p0[r]; }
#pragma unroll
        for (int r = 0; r < 8; ++r) { p1[r] = __expf(c1[r] - mnew); lsum += p1[r]; }
        lsum += __shfl_xor(lsum, 16, 32);
        lr = lr * alpha + lsum;
        mr = mnew;
#pragma unroll
        for (int t = 0; t < 4; ++t) ac[t] *= alpha;

#pragma unroll
        for (int j = 0; j < 4; ++j) {
          unsigned u0 = pk16(p0[2 * j], p0[2 * j + 1]);
          unsigned u1 = pk16(p1[2 * j], p1[2 * j + 1]);
          unsigned x0 = (unsigned)__shfl_xor((int)u0, 16, 32);
          unsigned x1 = (unsigned)__shfl_xor((int)u1, 16, 32);
          bp.u[j]     = low ? u0 : x1;
          bp.u[4 + j] = low ? x0 : u1;
        }
      }

      // ---- O^T += V^T * P^T: V tiles loaded once, used by both groups ----
#pragma unroll
      for (int t = 0; t < 4; ++t) {
        const f16* vrow = Vc + (t * 16 + lo16) * VROW;
        v16h av = ld2(vrow + 2 * sb, vrow + 2 * sb + 8);
        accA[t] = wmma16(av, bpa.v, accA[t]);
        accB[t] = wmma16(av, bpb.v, accB[t]);
      }
    }

    // next buffer must be complete before anyone reads it in iteration c+1
    if (wid < 2) __builtin_amdgcn_s_wait_tensorcnt(0);
    __syncthreads();
  }

  // ---- finalize: out[b, q, h, d] = accT[d, q] / l_run ----
  const int doff = low ? 0 : 8;
  {
    const float inv = 1.0f / lA;
    float* orow = out + (((size_t)(b * Ln + qa) * Hn + h) << 6);
#pragma unroll
    for (int t = 0; t < 4; ++t) {
      v8f sc = accA[t] * inv;
      *(f4*)(orow + t * 16 + doff)     = __builtin_shufflevector(sc, sc, 0, 1, 2, 3);
      *(f4*)(orow + t * 16 + doff + 4) = __builtin_shufflevector(sc, sc, 4, 5, 6, 7);
    }
  }
  {
    const float inv = 1.0f / lB;
    float* orow = out + (((size_t)(b * Ln + qb) * Hn + h) << 6);
#pragma unroll
    for (int t = 0; t < 4; ++t) {
      v8f sc = accB[t] * inv;
      *(f4*)(orow + t * 16 + doff)     = __builtin_shufflevector(sc, sc, 0, 1, 2, 3);
      *(f4*)(orow + t * 16 + doff + 4) = __builtin_shufflevector(sc, sc, 4, 5, 6, 7);
    }
  }
}

extern "C" void kernel_launch(void* const* d_in, const int* in_sizes, int n_in,
                              void* d_out, int out_size, void* d_ws, size_t ws_size,
                              hipStream_t stream) {
  const float* Q = (const float*)d_in[0];
  const float* K = (const float*)d_in[1];
  const float* V = (const float*)d_in[2];
  const float* W = (const float*)d_in[3];
  float* out = (float*)d_out;

  char* wsp = (char*)d_ws;
  f16* Qh = (f16*)(wsp);
  f16* Kh = (f16*)(wsp + (8u << 20));
  f16* Vt = (f16*)(wsp + (16u << 20));

  qw_kernel<<<(Bn * Ln * Hn) / 4, 256, 0, stream>>>(Q, W, Qh);
  kv_kernel<<<(Bn * Ln * Hn * En) / 256, 256, 0, stream>>>(K, V, Kh, Vt);
  fa_kernel<<<(Ln / 128) * Bn * Hn, 128, 0, stream>>>(Qh, Kh, Vt, out);
}
